// SpellerModel_34660386078825
// MI455X (gfx1250) — matrix operand order repaired and checked by reading, not compiled, not run on hardware
//
#include <hip/hip_runtime.h>
#include <hip/hip_bf16.h>
#include <math.h>

typedef __attribute__((ext_vector_type(16))) __bf16 v16bf;
typedef __attribute__((ext_vector_type(8)))  float  v8f;

namespace {
constexpr int B  = 64;    // batch
constexpr int H  = 512;   // hidden
constexpr int E  = 256;   // embed dim
constexpr int K  = 256;   // attention key dim
constexpr int S  = 400;   // source length
constexpr int T  = 200;   // target length
constexpr int V  = 34;    // vocab
constexpr int NG = 4 * H;     // 2048 gate columns
constexpr int KCAT = 1024;    // fused K for gate GEMMs ([x|h])
}

// ---------------------------------------------------------------------------
// WMMA A-tile (16x32 bf16) index for element (m, k) of an (M x Ktot) matrix,
// KT = Ktot/32.  Layout (contiguous): lanes 0-15 hold rows m, element e
// maps to K = e; lanes 16-31 hold K+16.
// ---------------------------------------------------------------------------
__device__ __forceinline__ int packedAIndex(int m, int k, int KT) {
  int mt = m >> 4, ml = m & 15;
  int kt = k >> 5, ki = k & 31;
  int e = ki & 15, lh = ki >> 4;
  return (((mt * KT + kt) * 32) + lh * 16 + ml) * 16 + e;
}

// ---------------------------------------------------------------------------
// Pack a (N x Ktot) row-major f32 weight pair [s0 | s1] (used transposed:
// B[k][n] = W[n][k]) into bf16 WMMA-B tiles:
// idx = ((nt*KT + kt)*32 + lane)*16 + e,
// k = kt*32 + (lane>=16 ? 16 : 0) + e, n = nt*16 + (lane&15).
// ---------------------------------------------------------------------------
__global__ void pack_b_kernel(const float* __restrict__ s0, const float* __restrict__ s1,
                              int k0, int Ktot, __bf16* __restrict__ dst, int total) {
  int tid = blockIdx.x * blockDim.x + threadIdx.x;
  if (tid >= total) return;
  int e    = tid & 15;
  int lane = (tid >> 4) & 31;
  int KT   = Ktot >> 5;
  int kt   = (tid >> 9) % KT;
  int nt   = (tid >> 9) / KT;
  int k = kt * 32 + ((lane >= 16) ? 16 : 0) + e;
  int n = nt * 16 + (lane & 15);
  float v = (k < k0) ? s0[(size_t)n * k0 + k]
                     : s1[(size_t)n * (Ktot - k0) + (k - k0)];
  dst[tid] = (__bf16)v;
}

// ---------------------------------------------------------------------------
// D(M x N) = A(M x Ktot) * B(Ktot x N) + bias, bf16 in, f32 accumulate.
// One wave per 16x16 output tile; A/B pre-packed in WMMA tile layouts.
// ---------------------------------------------------------------------------
__global__ void gemm_bf16_wmma(const __bf16* __restrict__ A, const __bf16* __restrict__ Bm,
                               const float* __restrict__ bias, float* __restrict__ D,
                               int Ntiles, int KT, int ldD, int numTiles) {
  int wave = (int)(threadIdx.x >> 5);
  int lane = (int)(threadIdx.x & 31);
  int tile = blockIdx.x * (int)(blockDim.x >> 5) + wave;
  if (tile >= numTiles) return;               // wave-uniform: EXEC stays all-1s
  int mt = tile / Ntiles;
  int nt = tile % Ntiles;
  const v16bf* pA = (const v16bf*)A  + (size_t)(mt * KT) * 32 + lane;
  const v16bf* pB = (const v16bf*)Bm + (size_t)(nt * KT) * 32 + lane;
  v8f acc = {};
  for (int kt = 0; kt < KT; ++kt) {
    v16bf a  = pA[(size_t)kt * 32];
    v16bf bb = pB[(size_t)kt * 32];
    acc = __builtin_amdgcn_wmma_f32_16x16x32_bf16(false, a, false, bb,
                                                  (short)0, acc, false, false);
  }
  int n = nt * 16 + (lane & 15);
  float bv = bias[n];
  int mbase = mt * 16 + ((lane >= 16) ? 8 : 0);
#pragma unroll
  for (int r = 0; r < 8; ++r)
    D[(size_t)(mbase + r) * ldD + n] = acc[r] + bv;
}

__device__ __forceinline__ float sigmoidf(float x) { return 1.0f / (1.0f + expf(-x)); }

// ---------------------------------------------------------------------------
// LSTM cell elementwise; scatter the new h (bf16) directly into the packed
// WMMA-A buffers of its two consumers:
//   dstA0 @ koff0 (KT0): the next GEMM within this step,
//   dstA1 @ koff1 (KT1): this layer's own gate GEMM next step.
// ---------------------------------------------------------------------------
__global__ void lstm_cell_pack_kernel(const float* __restrict__ gates,
                                      float* __restrict__ h, float* __restrict__ c,
                                      __bf16* __restrict__ dstA0, int koff0, int KT0,
                                      __bf16* __restrict__ dstA1, int koff1, int KT1) {
  int tid = blockIdx.x * blockDim.x + threadIdx.x;  // B*H
  if (tid >= B * H) return;
  int b = tid / H, j = tid % H;
  const float* g = gates + (size_t)b * NG;
  float ig = sigmoidf(g[j]);
  float fg = sigmoidf(g[H + j]);
  float gg = tanhf(g[2 * H + j]);
  float og = sigmoidf(g[3 * H + j]);
  float cn = fg * c[tid] + ig * gg;
  float hn = og * tanhf(cn);
  c[tid] = cn;
  h[tid] = hn;
  __bf16 hb = (__bf16)hn;
  dstA0[packedAIndex(b, koff0 + j, KT0)] = hb;
  dstA1[packedAIndex(b, koff1 + j, KT1)] = hb;
}

__global__ void add_bias_kernel(const float* __restrict__ a, const float* __restrict__ b,
                                float* __restrict__ d, int n) {
  int i = blockIdx.x * blockDim.x + threadIdx.x;
  if (i < n) d[i] = a[i] + b[i];
}

// ---------------------------------------------------------------------------
// Init: zero loss, broadcast h/c states (also into packed recurrent slots of
// pA1/pA2/pA3), compute ctx0 = hx1*Wq^T + bq, and seed pA1 with the step-0
// token embedding (token 0) and ctx0.  Leaves every packed buffer that a GEMM
// reads before a producer writes it fully initialized (d_ws is poisoned).
// ---------------------------------------------------------------------------
__global__ void init_kernel(const float* __restrict__ hx1, const float* __restrict__ cx1,
                            const float* __restrict__ hx2, const float* __restrict__ cx2,
                            const float* __restrict__ hx3, const float* __restrict__ cx3,
                            const float* __restrict__ Wq, const float* __restrict__ bq,
                            const float* __restrict__ emb,
                            float* __restrict__ h1, float* __restrict__ c1,
                            float* __restrict__ h2, float* __restrict__ c2,
                            float* __restrict__ h3, float* __restrict__ c3,
                            __bf16* __restrict__ pA1, __bf16* __restrict__ pA2,
                            __bf16* __restrict__ pA3, float* __restrict__ dout) {
  int tid = blockIdx.x * blockDim.x + threadIdx.x;
  if (tid >= B * H) return;
  if (tid == 0) dout[0] = 0.0f;
  int b = tid / H, j = tid % H;
  h1[tid] = hx1[j]; c1[tid] = cx1[j];
  h2[tid] = hx2[j]; c2[tid] = cx2[j];
  h3[tid] = hx3[j]; c3[tid] = cx3[j];
  pA1[packedAIndex(b, E + K + j, KCAT / 32)] = (__bf16)hx1[j];
  pA2[packedAIndex(b, H + j,     KCAT / 32)] = (__bf16)hx2[j];
  pA3[packedAIndex(b, H + j,     KCAT / 32)] = (__bf16)hx3[j];
  if (j < K) {
    float acc = bq[j];
    for (int k2 = 0; k2 < H; ++k2) acc += hx1[k2] * Wq[(size_t)j * H + k2];
    pA1[packedAIndex(b, E + j, KCAT / 32)] = (__bf16)acc;          // ctx0
    pA1[packedAIndex(b, j,     KCAT / 32)] = (__bf16)emb[j];        // token 0 row
  }
}

// ---------------------------------------------------------------------------
// Fused per-batch tail of a step: energy -> softmax (write attn_map row for
// b==0) -> masked renormalize -> context; pack new ctx and the *next* step's
// token embedding into pA1; then logits = h3*Wv^T+bv, log-softmax, NLL
// accumulation into dout[0] (t < T only).  One block of 256 per batch element.
// ---------------------------------------------------------------------------
__global__ void attention_fused_kernel(const float* __restrict__ q, const float* __restrict__ key,
                                       const float* __restrict__ val, const float* __restrict__ amask,
                                       const float* __restrict__ emb, const int* __restrict__ target,
                                       const float* __restrict__ h3, const float* __restrict__ Wv,
                                       const float* __restrict__ bvv, const float* __restrict__ tmask,
                                       __bf16* __restrict__ pA1, int t, float* __restrict__ dout) {
  __shared__ float qs[K];
  __shared__ float at[S];
  __shared__ float an[S];
  __shared__ float red[256];
  __shared__ float lg[V];
  int b = blockIdx.x, tid = threadIdx.x;
  qs[tid] = q[(size_t)b * K + tid];
  __syncthreads();
  float lmax = -1e30f;
  for (int s = tid; s < S; s += 256) {
    const float* kp = key + ((size_t)b * S + s) * K;
    float acc = 0.0f;
    for (int k2 = 0; k2 < K; ++k2) acc += qs[k2] * kp[k2];
    at[s] = acc;
    lmax = fmaxf(lmax, acc);
  }
  red[tid] = lmax; __syncthreads();
  for (int st = 128; st; st >>= 1) { if (tid < st) red[tid] = fmaxf(red[tid], red[tid + st]); __syncthreads(); }
  float mx = red[0]; __syncthreads();
  float lsum = 0.0f;
  for (int s = tid; s < S; s += 256) { float e = expf(at[s] - mx); at[s] = e; lsum += e; }
  red[tid] = lsum; __syncthreads();
  for (int st = 128; st; st >>= 1) { if (tid < st) red[tid] += red[tid + st]; __syncthreads(); }
  float inv = 1.0f / red[0]; __syncthreads();
  float labs = 0.0f;
  for (int s = tid; s < S; s += 256) {
    float a = at[s] * inv;                        // softmax output (pre-mask)
    if (b == 0) dout[1 + (size_t)t * S + s] = a;  // attn_map row
    float am = a * amask[(size_t)b * S + s];
    an[s] = am;
    labs += fabsf(am);
  }
  red[tid] = labs; __syncthreads();
  for (int st = 128; st; st >>= 1) { if (tid < st) red[tid] += red[tid + st]; __syncthreads(); }
  float dinv = 1.0f / fmaxf(red[0], 1e-12f);
  __syncthreads();
  {
    int k2 = tid;  // 0..255 == K
    float acc = 0.0f;
    for (int s = 0; s < S; ++s) acc += an[s] * val[((size_t)b * S + s) * K + k2];
    float cv = acc * dinv;
    pA1[packedAIndex(b, E + k2, KCAT / 32)] = (__bf16)cv;   // ctx for step t+1
    if (t < T) {                                            // token for step t+1
      int row = target[t * B + b];
      pA1[packedAIndex(b, k2, KCAT / 32)] = (__bf16)emb[(size_t)row * E + k2];
    }
  }
  // ---- logits / loss (t < T only) ----
  if (t < T) {
    if (tid < V) {
      const float* w  = Wv + (size_t)tid * H;
      const float* hp = h3 + (size_t)b * H;
      float acc = bvv[tid];
      for (int k2 = 0; k2 < H; ++k2) acc += hp[k2] * w[k2];
      lg[tid] = acc;
    }
    __syncthreads();
    if (tid == 0) {
      float mx2 = lg[0];
      for (int v = 1; v < V; ++v) mx2 = fmaxf(mx2, lg[v]);
      float sum = 0.0f;
      for (int v = 0; v < V; ++v) sum += expf(lg[v] - mx2);
      int tg = target[t * B + b];
      float nll = -(lg[tg] - mx2 - logf(sum));
      atomicAdd(dout, nll * tmask[t * B + b] * (1.0f / (float)B));
    }
  }
}

extern "C" void kernel_launch(void* const* d_in, const int* in_sizes, int n_in,
                              void* d_out, int out_size, void* d_ws, size_t ws_size,
                              hipStream_t stream) {
  (void)in_sizes; (void)n_in; (void)out_size; (void)ws_size;
  const int*   target = (const int*)  d_in[0];
  const float* tmask  = (const float*)d_in[1];
  const float* akey   = (const float*)d_in[2];
  const float* aval   = (const float*)d_in[3];
  const float* amask  = (const float*)d_in[4];
  const float* emb    = (const float*)d_in[5];
  const float* Wih[3] = { (const float*)d_in[6],  (const float*)d_in[10], (const float*)d_in[14] };
  const float* Whh[3] = { (const float*)d_in[7],  (const float*)d_in[11], (const float*)d_in[15] };
  const float* bih[3] = { (const float*)d_in[8],  (const float*)d_in[12], (const float*)d_in[16] };
  const float* bhh[3] = { (const float*)d_in[9],  (const float*)d_in[13], (const float*)d_in[17] };
  const float* Wq  = (const float*)d_in[18];
  const float* bq  = (const float*)d_in[19];
  const float* Wv  = (const float*)d_in[20];
  const float* bvv = (const float*)d_in[21];
  const float* hx[3] = { (const float*)d_in[22], (const float*)d_in[24], (const float*)d_in[26] };
  const float* cx[3] = { (const float*)d_in[23], (const float*)d_in[25], (const float*)d_in[27] };
  float* dout = (float*)d_out;

  char* wsb = (char*)d_ws;
  size_t off = 0;
  auto carve = [&](size_t bytes) -> void* {
    void* p = wsb + off;
    off = (off + bytes + 255) & ~(size_t)255;
    return p;
  };
  __bf16* pB[3]; for (int l = 0; l < 3; ++l) pB[l] = (__bf16*)carve((size_t)NG * KCAT * 2);
  __bf16* pBq = (__bf16*)carve((size_t)K * H * 2);
  float* bc[3]; for (int l = 0; l < 3; ++l) bc[l] = (float*)carve((size_t)NG * 4);
  float* h[3]; float* c[3];
  for (int l = 0; l < 3; ++l) { h[l] = (float*)carve((size_t)B * H * 4); c[l] = (float*)carve((size_t)B * H * 4); }
  float*  qb    = (float*) carve((size_t)B * K * 4);
  float*  gates = (float*) carve((size_t)B * NG * 4);
  __bf16* pA1   = (__bf16*)carve((size_t)B * KCAT * 2);  // [emb | ctx | h1]
  __bf16* pA2   = (__bf16*)carve((size_t)B * KCAT * 2);  // [h1 | h2]
  __bf16* pA3   = (__bf16*)carve((size_t)B * KCAT * 2);  // [h2 | h3]
  __bf16* pAq   = (__bf16*)carve((size_t)B * H * 2);     // [h3]

  // --- one-time weight packing (bf16, WMMA-B layout) + bias fuse ---
  const int totB = NG * KCAT;
  for (int l = 0; l < 3; ++l) {
    pack_b_kernel<<<(totB + 255) / 256, 256, 0, stream>>>(Wih[l], Whh[l], H, KCAT, pB[l], totB);
    add_bias_kernel<<<(NG + 255) / 256, 256, 0, stream>>>(bih[l], bhh[l], bc[l], NG);
  }
  const int totBq = K * H;
  pack_b_kernel<<<(totBq + 255) / 256, 256, 0, stream>>>(Wq, Wq, H, H, pBq, totBq);

  init_kernel<<<(B * H + 255) / 256, 256, 0, stream>>>(
      hx[0], cx[0], hx[1], cx[1], hx[2], cx[2], Wq, bq, emb,
      h[0], c[0], h[1], c[1], h[2], c[2], pA1, pA2, pA3, dout);

  const int gatesTiles = (B / 16) * (NG / 16);  // 512 tiles -> 64 blocks x 8 waves
  const int qTiles     = (B / 16) * (K / 16);   // 64 tiles  -> 8 blocks

  for (int t = 0; t <= T; ++t) {
    // layer 1: gates = [emb|ctx|h1] * [Wih1|Whh1]^T
    gemm_bf16_wmma<<<gatesTiles / 8, 256, 0, stream>>>(
        pA1, pB[0], bc[0], gates, NG / 16, KCAT / 32, NG, gatesTiles);
    lstm_cell_pack_kernel<<<(B * H + 255) / 256, 256, 0, stream>>>(
        gates, h[0], c[0], pA2, 0, KCAT / 32, pA1, E + K, KCAT / 32);

    // layer 2: gates = [h1|h2] * [Wih2|Whh2]^T
    gemm_bf16_wmma<<<gatesTiles / 8, 256, 0, stream>>>(
        pA2, pB[1], bc[1], gates, NG / 16, KCAT / 32, NG, gatesTiles);
    lstm_cell_pack_kernel<<<(B * H + 255) / 256, 256, 0, stream>>>(
        gates, h[1], c[1], pA3, 0, KCAT / 32, pA2, H, KCAT / 32);

    // layer 3: gates = [h2|h3] * [Wih3|Whh3]^T
    gemm_bf16_wmma<<<gatesTiles / 8, 256, 0, stream>>>(
        pA3, pB[2], bc[2], gates, NG / 16, KCAT / 32, NG, gatesTiles);
    lstm_cell_pack_kernel<<<(B * H + 255) / 256, 256, 0, stream>>>(
        gates, h[2], c[2], pAq, 0, H / 32, pA3, H, KCAT / 32);

    // q = h3 * Wq^T + bq  (WMMA)
    gemm_bf16_wmma<<<qTiles / 8, 256, 0, stream>>>(
        pAq, pBq, bq, qb, K / 16, H / 32, K, qTiles);

    // attention + ctx/emb packing for t+1 + logits/loss
    attention_fused_kernel<<<B, 256, 0, stream>>>(
        qb, akey, aval, amask, emb, target, h[2], Wv, bvv, tmask, pA1, t, dout);
  }
}
